// mTAN_enc_41300405518549
// MI455X (gfx1250) — matrix operand order repaired and checked by reading, not compiled, not run on hardware
//
#include <hip/hip_runtime.h>
#include <hip/hip_bf16.h>
#include <math.h>

// ---------------------------------------------------------------------------
// mTAN encoder for MI455X (gfx1250, wave32, WMMA f16 16x16x32).
// Pipeline:
//   1) f32->f16 weight converts + x transpose (xT[b][d][t] f16)
//   2) temb+projection GEMM (WMMA): key_e/q_e built in LDS, x Wk^T / Wq^T
//   3) flash attention (WMMA): S^T = K*Q^T tiles, online softmax over T,
//      O^T = X^T * P accumulated via WMMA, P built from S^T frag by shfl_xor(16)
//   4) hidden = att_cat @ Wo^T (WMMA), stored channel-major [B,64,256]
//   5) CNN tail: conv1d / two-pass BN stats / bn+act / pools / res / 3 FCs (VALU)
// ---------------------------------------------------------------------------

typedef __attribute__((ext_vector_type(16))) _Float16 v16h;
typedef __attribute__((ext_vector_type(8)))  float    v8f;

#define NB   32
#define NT   2048
#define NLQ  256
#define NDIM 16
#define NE   128
#define NHD  2
#define NEK  64
#define NNH  64
#define NCL2 128

static __device__ __forceinline__ v16h load_frag_a(const _Float16* base, int ld) {
  // A 16x32 f16 frag: lane m=lane%16, g=lane/16; elem j -> k=g*16+j
  int lane = threadIdx.x & 31;
  int m = lane & 15, g = lane >> 4;
  union { v16h v; float4 q[2]; } u;
  u.q[0] = *(const float4*)(base + m * ld + g * 16);
  u.q[1] = *(const float4*)(base + m * ld + g * 16 + 8);
  return u.v;
}

static __device__ __forceinline__ v16h load_frag_b(const _Float16* base, int ld) {
  // B 32x16 f16 frag from row-major [N][K] source: lane n=lane%16 holds k=g*16+j of column n
  int lane = threadIdx.x & 31;
  int n = lane & 15, g = lane >> 4;
  const _Float16* p = base + n * ld + g * 16;
  union { v16h v; float4 q[2]; } u;
  u.q[0] = *(const float4*)p;
  u.q[1] = *(const float4*)(p + 8);
  return u.v;
}

static __device__ __forceinline__ v8f wmma16(v16h a, v16h b, v8f c) {
  return __builtin_amdgcn_wmma_f32_16x16x32_f16(false, a, false, b, (short)0, c, false, false);
}

// --------------------------- prep kernels ---------------------------------

__global__ void k_f32_to_f16(const float* __restrict__ in, _Float16* __restrict__ out, int n) {
  int i = blockIdx.x * blockDim.x + threadIdx.x;
  if (i < n) out[i] = (_Float16)in[i];
}

__global__ void k_transpose_x(const float* __restrict__ x, _Float16* __restrict__ xT) {
  int i = blockIdx.x * blockDim.x + threadIdx.x;          // over B*T*16
  if (i >= NB * NT * NDIM) return;
  int d = i & 15; int t = (i >> 4) & (NT - 1); int b = i >> 15;
  xT[(b * NDIM + d) * NT + t] = (_Float16)x[i];
}

// ----------------- time-embedding + projection GEMM (WMMA) ----------------
// out[((row/Tp)*2 + h)*Tp + row%Tp]*64 + ek  where col j = h*64+ek
__global__ void k_temb_proj(const float* __restrict__ times,
                            const float* __restrict__ w_per, const float* __restrict__ b_per,
                            const float* __restrict__ w_lin, const float* __restrict__ b_lin,
                            const _Float16* __restrict__ W, const float* __restrict__ bias,
                            _Float16* __restrict__ out, int Tp) {
  __shared__ _Float16 ke[64 * NE];                         // 16 KB
  int row0 = blockIdx.x * 64;
  int tid = threadIdx.x;
  float wl = w_lin[0], bl = b_lin[0];
  for (int i = tid; i < 64 * NE; i += 128) {
    int r = i >> 7, e = i & 127;
    float t = times[row0 + r];
    float v = (e == 0) ? (t * wl + bl) : __sinf(t * w_per[e - 1] + b_per[e - 1]);
    ke[i] = (_Float16)v;
  }
  __syncthreads();
  int wave = tid >> 5;
  int lane = tid & 31, n16 = lane & 15, g = lane >> 4;
  const _Float16* abase = &ke[wave * 16 * NE];
  for (int nt = 0; nt < 8; ++nt) {
    float bv = bias[nt * 16 + n16];
    v8f c = {bv, bv, bv, bv, bv, bv, bv, bv};
#pragma unroll
    for (int k0 = 0; k0 < NE; k0 += 32) {
      v16h a = load_frag_a(abase + k0, NE);
      v16h b = load_frag_b(W + (nt * 16) * NE + k0, NE);
      c = wmma16(a, b, c);
    }
    int j = nt * 16 + n16, h = j >> 6, ek = j & 63;
#pragma unroll
    for (int r = 0; r < 8; ++r) {
      int row = row0 + wave * 16 + r + 8 * g;
      int bb = row / Tp, t = row % Tp;
      out[(((bb * NHD + h) * Tp + t)) * NEK + ek] = (_Float16)c[r];
    }
  }
}

// ------------------------- flash attention (WMMA) -------------------------
__global__ void k_flash_attn(const _Float16* __restrict__ qh,   // [H][LQ][64]
                             const _Float16* __restrict__ kh,   // [B][H][T][64]
                             const _Float16* __restrict__ xT,   // [B][16][T]
                             _Float16* __restrict__ att) {      // [B][LQ][32]
  int wid  = blockIdx.x * 4 + (threadIdx.x >> 5);
  int lane = threadIdx.x & 31, n16 = lane & 15, g = lane >> 4;
  int b = wid >> 5, rem = wid & 31, h = rem >> 4, lqt = rem & 15;

  const _Float16* qbase = qh + (h * NLQ + lqt * 16) * NEK;
  v16h bq0 = load_frag_b(qbase, NEK);                      // ek 0..31
  v16h bq1 = load_frag_b(qbase + 32, NEK);                 // ek 32..63
  const _Float16* kbase = kh + (size_t)((b * NHD + h) * NT) * NEK;
  const _Float16* xbase = xT + (size_t)(b * NDIM) * NT;

  v8f O = {};
  float mrun = -__builtin_inff(), lrun = 0.f;

  for (int t0 = 0; t0 < NT; t0 += 32) {
    if (t0 + 32 < NT) __builtin_prefetch(kbase + (size_t)(t0 + 32) * NEK, 0, 1);
    v8f z = {};
    // S^T tiles: rows t, cols lq
    v8f s0 = wmma16(load_frag_a(kbase + (size_t)t0 * NEK, NEK), bq0,
             wmma16(load_frag_a(kbase + (size_t)t0 * NEK + 32, NEK), bq1, z));
    v8f s1 = wmma16(load_frag_a(kbase + (size_t)(t0 + 16) * NEK, NEK), bq0,
             wmma16(load_frag_a(kbase + (size_t)(t0 + 16) * NEK + 32, NEK), bq1, z));
    s0 = s0 * 0.125f;                                      // 1/sqrt(64)
    s1 = s1 * 0.125f;

    float mloc = -__builtin_inff();
#pragma unroll
    for (int r = 0; r < 8; ++r) mloc = fmaxf(mloc, fmaxf(s0[r], s1[r]));
    mloc = fmaxf(mloc, __shfl_xor(mloc, 16, 32));
    float mnew  = fmaxf(mrun, mloc);
    float alpha = __expf(mrun - mnew);

    float p0[8], p1[8], ls = 0.f;
#pragma unroll
    for (int r = 0; r < 8; ++r) {
      p0[r] = __expf(s0[r] - mnew);
      p1[r] = __expf(s1[r] - mnew);
      ls += p0[r] + p1[r];
    }
    ls += __shfl_xor(ls, 16, 32);
    lrun = lrun * alpha + ls;
    mrun = mnew;
#pragma unroll
    for (int r = 0; r < 8; ++r) O[r] *= alpha;

    // Build P B-fragment [k=t(32) x n=lq(16)] from the two S^T D-fragments.
    union { v16h v; _Float16 e[16]; } up;
#pragma unroll
    for (int r = 0; r < 8; ++r) {
      float send = g ? p0[r] : p1[r];
      float recv = __shfl_xor(send, 16, 32);
      up.e[r]     = (_Float16)(g ? recv : p0[r]);          // k = g*16 + r
      up.e[8 + r] = (_Float16)(g ? p1[r] : recv);          // k = g*16 + 8 + r
    }
    v16h aX = load_frag_a(xbase + t0, NT);                 // X^T [16d x 32t]
    O = wmma16(aX, up.v, O);                               // att^T accumulate
  }

  float inv = 1.f / lrun;
#pragma unroll
  for (int r = 0; r < 8; ++r) {
    int m = r + 8 * g;                                     // d index
    att[((size_t)(b * NLQ + lqt * 16 + n16)) * 32 + h * NDIM + m] = (_Float16)(O[r] * inv);
  }
}

// ----------------- hidden = att_cat @ Wo^T, stored [B,64,LQ] --------------
__global__ void k_hidden_gemm(const _Float16* __restrict__ att,  // [B*LQ][32]
                              const _Float16* __restrict__ wo,   // [64][32]
                              const float* __restrict__ bo,
                              float* __restrict__ hid) {         // [B][64][LQ]
  int wid  = blockIdx.x * 4 + (threadIdx.x >> 5);
  int lane = threadIdx.x & 31, n16 = lane & 15, g = lane >> 4;
  int mt = wid >> 2, ntile = wid & 3;
  v16h a = load_frag_a(att + (size_t)mt * 16 * 32, 32);
  v16h b = load_frag_b(wo + ntile * 16 * 32, 32);
  float bv = bo[ntile * 16 + n16];
  v8f c = {bv, bv, bv, bv, bv, bv, bv, bv};
  c = wmma16(a, b, c);
#pragma unroll
  for (int r = 0; r < 8; ++r) {
    int row = mt * 16 + r + 8 * g;
    int bb = row >> 8, lq = row & 255, ch = ntile * 16 + n16;
    hid[((size_t)(bb * NNH + ch)) * NLQ + lq] = c[r];
  }
}

// ------------------------------ CNN tail ----------------------------------

__global__ void k_conv1d(const float* __restrict__ in, const float* __restrict__ W,
                         const float* __restrict__ bias, float* __restrict__ out,
                         int Cin, int Cout, int Lin, int Lout, int pad, int total) {
  int idx = blockIdx.x * blockDim.x + threadIdx.x;
  if (idx >= total) return;
  int l = idx % Lout; int t = idx / Lout; int co = t % Cout; int b = t / Cout;
  float acc = bias[co];
  for (int ci = 0; ci < Cin; ++ci) {
    const float* ip = in + (size_t)(b * Cin + ci) * Lin;
    const float* wp = W + (size_t)(co * Cin + ci) * 3;
#pragma unroll
    for (int k = 0; k < 3; ++k) {
      int pos = l + k - pad;
      if (pos >= 0 && pos < Lin) acc += ip[pos] * wp[k];
    }
  }
  out[idx] = acc;
}

__global__ void k_chan_stats(const float* __restrict__ in, float* __restrict__ stats,
                             int C, int L) {
  int c = blockIdx.x;
  __shared__ float ssum[256], ssq[256];
  float s = 0.f, q = 0.f;
  for (int i = threadIdx.x; i < NB * L; i += 256) {
    int b = i / L, l = i % L;
    float v = in[(size_t)(b * C + c) * L + l];
    s += v; q += v * v;
  }
  ssum[threadIdx.x] = s; ssq[threadIdx.x] = q;
  __syncthreads();
  for (int st = 128; st > 0; st >>= 1) {
    if (threadIdx.x < st) { ssum[threadIdx.x] += ssum[threadIdx.x + st];
                            ssq[threadIdx.x]  += ssq[threadIdx.x + st]; }
    __syncthreads();
  }
  if (threadIdx.x == 0) {
    float N = (float)(NB * L);
    float m = ssum[0] / N;
    stats[2 * c] = m;
    stats[2 * c + 1] = ssq[0] / N - m * m;                 // biased var
  }
}

__global__ void k_bn_act(float* __restrict__ v, const float* __restrict__ stats,
                         const float* __restrict__ gam, const float* __restrict__ bet,
                         int C, int L, int total, int act) {
  int idx = blockIdx.x * blockDim.x + threadIdx.x;
  if (idx >= total) return;
  int c = (idx / L) % C;
  float m = stats[2 * c], var = stats[2 * c + 1];
  float x = (v[idx] - m) * rsqrtf(var + 1e-5f) * gam[c] + bet[c];
  if (act == 1)      x = fmaxf(x, 0.f);
  else if (act == 2) x = (x > 0.f) ? x : 0.01f * x;
  v[idx] = x;
}

__global__ void k_pool2(const float* __restrict__ in, float* __restrict__ out,
                        int Lin, int Lout, int total, int isMax) {
  int idx = blockIdx.x * blockDim.x + threadIdx.x;
  if (idx >= total) return;
  int l = idx % Lout; int rest = idx / Lout;
  float a = in[(size_t)rest * Lin + 2 * l];
  float b = in[(size_t)rest * Lin + 2 * l + 1];
  out[idx] = isMax ? fmaxf(a, b) : 0.5f * (a + b);
}

__global__ void k_add_lrelu(const float* __restrict__ a, float* __restrict__ r, int total) {
  int idx = blockIdx.x * blockDim.x + threadIdx.x;
  if (idx >= total) return;
  float x = a[idx] + r[idx];
  r[idx] = (x > 0.f) ? x : 0.01f * x;
}

__global__ void k_fc_relu(const float* __restrict__ in, const float* __restrict__ W,
                          const float* __restrict__ bias, float* __restrict__ out,
                          int K, int N, int total) {
  int idx = blockIdx.x * blockDim.x + threadIdx.x;
  if (idx >= total) return;
  int n = idx % N, b = idx / N;
  const float* ip = in + (size_t)b * K;
  const float* wp = W + (size_t)n * K;
  float acc = bias[n];
  for (int k = 0; k < K; ++k) acc += ip[k] * wp[k];
  out[idx] = fmaxf(acc, 0.f);
}

// ------------------------------ launcher ----------------------------------

extern "C" void kernel_launch(void* const* d_in, const int* in_sizes, int n_in,
                              void* d_out, int out_size, void* d_ws, size_t ws_size,
                              hipStream_t stream) {
  (void)in_sizes; (void)n_in; (void)out_size; (void)ws_size;
  const float* x      = (const float*)d_in[0];
  const float* tsteps = (const float*)d_in[1];
  const float* query  = (const float*)d_in[2];
  const float* w_per  = (const float*)d_in[3];
  const float* b_per  = (const float*)d_in[4];
  const float* w_lin  = (const float*)d_in[5];
  const float* b_lin  = (const float*)d_in[6];
  const float* Wq = (const float*)d_in[7];   const float* bq = (const float*)d_in[8];
  const float* Wk = (const float*)d_in[9];   const float* bk = (const float*)d_in[10];
  const float* Wo = (const float*)d_in[11];  const float* bo = (const float*)d_in[12];
  const float* c1w = (const float*)d_in[13]; const float* c1b = (const float*)d_in[14];
  const float* g1  = (const float*)d_in[15]; const float* be1 = (const float*)d_in[16];
  const float* r1c1w = (const float*)d_in[17]; const float* r1c1b = (const float*)d_in[18];
  const float* r1g1  = (const float*)d_in[19]; const float* r1be1 = (const float*)d_in[20];
  const float* r1c2w = (const float*)d_in[21]; const float* r1c2b = (const float*)d_in[22];
  const float* r1g2  = (const float*)d_in[23]; const float* r1be2 = (const float*)d_in[24];
  const float* r2c1w = (const float*)d_in[25]; const float* r2c1b = (const float*)d_in[26];
  const float* r2g1  = (const float*)d_in[27]; const float* r2be1 = (const float*)d_in[28];
  const float* r2c2w = (const float*)d_in[29]; const float* r2c2b = (const float*)d_in[30];
  const float* r2g2  = (const float*)d_in[31]; const float* r2be2 = (const float*)d_in[32];
  const float* c2w = (const float*)d_in[33]; const float* c2b = (const float*)d_in[34];
  const float* g2  = (const float*)d_in[35]; const float* be2 = (const float*)d_in[36];
  const float* fc1w = (const float*)d_in[37]; const float* fc1b = (const float*)d_in[38];
  const float* fc2w = (const float*)d_in[39]; const float* fc2b = (const float*)d_in[40];
  const float* fc3w = (const float*)d_in[41]; const float* fc3b = (const float*)d_in[42];

  char* ws = (char*)d_ws;
  size_t off = 0;
  auto take = [&](size_t bytes) -> char* {
    char* p = ws + off; off += (bytes + 255) & ~(size_t)255; return p;
  };
  _Float16* qh   = (_Float16*)take((size_t)NHD * NLQ * NEK * 2);
  _Float16* wqh  = (_Float16*)take((size_t)NE * NE * 2);
  _Float16* wkh  = (_Float16*)take((size_t)NE * NE * 2);
  _Float16* woh  = (_Float16*)take((size_t)NNH * 32 * 2);
  _Float16* khp  = (_Float16*)take((size_t)NB * NHD * NT * NEK * 2);  // 16 MB
  _Float16* xT   = (_Float16*)take((size_t)NB * NDIM * NT * 2);
  _Float16* attc = (_Float16*)take((size_t)NB * NLQ * 32 * 2);
  float* hid = (float*)take((size_t)NB * NNH * NLQ * 4);
  float* C1  = (float*)take((size_t)NB * NNH * 254 * 4);
  float* P1  = (float*)take((size_t)NB * NNH * 127 * 4);
  float* T1  = (float*)take((size_t)NB * NNH * 127 * 4);
  float* T2  = (float*)take((size_t)NB * NNH * 127 * 4);
  float* AV  = (float*)take((size_t)NB * NNH * 63 * 4);
  float* C2  = (float*)take((size_t)NB * NCL2 * 61 * 4);
  float* P2  = (float*)take((size_t)NB * NCL2 * 30 * 4);
  float* F1  = (float*)take((size_t)NB * 128 * 4);
  float* F2  = (float*)take((size_t)NB * 64 * 4);
  float* ST  = (float*)take((size_t)NCL2 * 2 * 4);

  // 1) converts + x transpose
  k_f32_to_f16<<<64, 256, 0, stream>>>(Wq, wqh, NE * NE);
  k_f32_to_f16<<<64, 256, 0, stream>>>(Wk, wkh, NE * NE);
  k_f32_to_f16<<<8, 256, 0, stream>>>(Wo, woh, NNH * 32);
  k_transpose_x<<<(NB * NT * NDIM) / 256, 256, 0, stream>>>(x, xT);

  // 2) projections (WMMA)
  k_temb_proj<<<NLQ / 64, 128, 0, stream>>>(query, w_per, b_per, w_lin, b_lin,
                                            wqh, bq, qh, NLQ);
  k_temb_proj<<<(NB * NT) / 64, 128, 0, stream>>>(tsteps, w_per, b_per, w_lin, b_lin,
                                                  wkh, bk, khp, NT);

  // 3) fused flash attention (WMMA)
  k_flash_attn<<<(NB * NHD * (NLQ / 16)) / 4, 128, 0, stream>>>(qh, khp, xT, attc);

  // 4) hidden projection (WMMA), transposed to [B,64,LQ]
  k_hidden_gemm<<<((NB * NLQ / 16) * 4) / 4, 128, 0, stream>>>(attc, woh, bo, hid);

  // 5) CNN tail
  int tot;
  tot = NB * NNH * 254;
  k_conv1d<<<(tot + 255) / 256, 256, 0, stream>>>(hid, c1w, c1b, C1, NNH, NNH, 256, 254, 0, tot);
  k_chan_stats<<<NNH, 256, 0, stream>>>(C1, ST, NNH, 254);
  k_bn_act<<<(tot + 255) / 256, 256, 0, stream>>>(C1, ST, g1, be1, NNH, 254, tot, 1);
  tot = NB * NNH * 127;
  k_pool2<<<(tot + 255) / 256, 256, 0, stream>>>(C1, P1, 254, 127, tot, 1);

  // residual block 1
  k_conv1d<<<(tot + 255) / 256, 256, 0, stream>>>(P1, r1c1w, r1c1b, T1, NNH, NNH, 127, 127, 1, tot);
  k_chan_stats<<<NNH, 256, 0, stream>>>(T1, ST, NNH, 127);
  k_bn_act<<<(tot + 255) / 256, 256, 0, stream>>>(T1, ST, r1g1, r1be1, NNH, 127, tot, 2);
  k_conv1d<<<(tot + 255) / 256, 256, 0, stream>>>(T1, r1c2w, r1c2b, T2, NNH, NNH, 127, 127, 1, tot);
  k_chan_stats<<<NNH, 256, 0, stream>>>(T2, ST, NNH, 127);
  k_bn_act<<<(tot + 255) / 256, 256, 0, stream>>>(T2, ST, r1g2, r1be2, NNH, 127, tot, 2);
  k_add_lrelu<<<(tot + 255) / 256, 256, 0, stream>>>(T2, P1, tot);

  // residual block 2
  k_conv1d<<<(tot + 255) / 256, 256, 0, stream>>>(P1, r2c1w, r2c1b, T1, NNH, NNH, 127, 127, 1, tot);
  k_chan_stats<<<NNH, 256, 0, stream>>>(T1, ST, NNH, 127);
  k_bn_act<<<(tot + 255) / 256, 256, 0, stream>>>(T1, ST, r2g1, r2be1, NNH, 127, tot, 2);
  k_conv1d<<<(tot + 255) / 256, 256, 0, stream>>>(T1, r2c2w, r2c2b, T2, NNH, NNH, 127, 127, 1, tot);
  k_chan_stats<<<NNH, 256, 0, stream>>>(T2, ST, NNH, 127);
  k_bn_act<<<(tot + 255) / 256, 256, 0, stream>>>(T2, ST, r2g2, r2be2, NNH, 127, tot, 2);
  k_add_lrelu<<<(tot + 255) / 256, 256, 0, stream>>>(T2, P1, tot);

  tot = NB * NNH * 63;
  k_pool2<<<(tot + 255) / 256, 256, 0, stream>>>(P1, AV, 127, 63, tot, 0);

  tot = NB * NCL2 * 61;
  k_conv1d<<<(tot + 255) / 256, 256, 0, stream>>>(AV, c2w, c2b, C2, NNH, NCL2, 63, 61, 0, tot);
  k_chan_stats<<<NCL2, 256, 0, stream>>>(C2, ST, NCL2, 61);
  k_bn_act<<<(tot + 255) / 256, 256, 0, stream>>>(C2, ST, g2, be2, NCL2, 61, tot, 1);
  tot = NB * NCL2 * 30;
  k_pool2<<<(tot + 255) / 256, 256, 0, stream>>>(C2, P2, 61, 30, tot, 1);

  // FC head (flatten of [B,128,30] row-major == reference reshape)
  k_fc_relu<<<16, 256, 0, stream>>>(P2, fc1w, fc1b, F1, NCL2 * 30, 128, NB * 128);
  k_fc_relu<<<8, 256, 0, stream>>>(F1, fc2w, fc2b, F2, 128, 64, NB * 64);
  k_fc_relu<<<1, 256, 0, stream>>>(F2, fc3w, fc3b, (float*)d_out, 64, 6, NB * 6);
}
